// EncoderBlock_25357486915895
// MI455X (gfx1250) — matrix operand order repaired and checked by reading, not compiled, not run on hardware
//
#include <hip/hip_runtime.h>
#include <hip/hip_bf16.h>
#include <math.h>

// Problem dims
#define B_DIM 8
#define N_DIM 512
#define DN    128            // d_node / head dim
#define DE    16             // d_edge / num heads
#define PROJ  (DN*DE)        // 2048
#define ROWS  (B_DIM*N_DIM)  // 4096
#define EPS   1e-5f

typedef __bf16 bf16;
typedef bf16  v16bf __attribute__((ext_vector_type(16)));
typedef bf16  v8bf  __attribute__((ext_vector_type(8)));
typedef float v8f   __attribute__((ext_vector_type(8)));

__device__ __forceinline__ float swishf(float y) {
    return y * (1.0f / (1.0f + __expf(-y)));
}

// -------------------------------------------------------------------------
// Kernel 0: transpose + convert weights W[K,C] f32 -> Wt[C,K] bf16 so that
// GEMM B-fragments become two contiguous b128 loads per lane.
// C must be a power of two (2048 or 128); cshift = log2(C).
// -------------------------------------------------------------------------
__global__ __launch_bounds__(256) void transpose_w_kernel(
    const float* __restrict__ W, bf16* __restrict__ Wt, int cshift, int K)
{
    size_t idx = (size_t)blockIdx.x * 256 + threadIdx.x;   // over K*C
    int cmask = (1 << cshift) - 1;
    int k = (int)(idx >> cshift);
    int c = (int)(idx & cmask);
    Wt[(size_t)c * K + k] = (bf16)W[idx];
}

// -------------------------------------------------------------------------
// Kernel 1: fused edge pipeline. One thread per edge (b,n,m).
//   fe = LN-swish(f_edges) @ ep_W + ep_bias              (16x16)
//   score_ws[b,h,n,m] = adj==0 ? -1e9 : fe[h]   (bf16, head-major layout)
//   edges_out = LN-swish(fe) @ eo_W + eo_bias + f_edges
// -------------------------------------------------------------------------
__global__ __launch_bounds__(256) void edge_kernel(
    const float* __restrict__ f_edges, const int* __restrict__ adj,
    const float* __restrict__ ep_g, const float* __restrict__ ep_b,
    const float* __restrict__ ep_W, const float* __restrict__ ep_bias,
    const float* __restrict__ eo_g, const float* __restrict__ eo_b,
    const float* __restrict__ eo_W, const float* __restrict__ eo_bias,
    bf16* __restrict__ score_ws, float* __restrict__ edges_out)
{
    __shared__ float sW[256], sWo[256];
    __shared__ float sg[16], sb[16], sbias[16], sog[16], sob[16], sobias[16];
    int tid = threadIdx.x;
    sW[tid]  = ep_W[tid];
    sWo[tid] = eo_W[tid];
    if (tid < 16) {
        sg[tid] = ep_g[tid];  sb[tid] = ep_b[tid];  sbias[tid] = ep_bias[tid];
        sog[tid] = eo_g[tid]; sob[tid] = eo_b[tid]; sobias[tid] = eo_bias[tid];
    }
    __syncthreads();

    size_t e = (size_t)blockIdx.x * 256 + tid;      // < 8*512*512
    int b = (int)(e >> 18);
    int rem = (int)(e & ((1 << 18) - 1));
    int n = rem >> 9, m = rem & 511;

    const float4* xp = (const float4*)(f_edges + e * 16);
    float xv[16];
#pragma unroll
    for (int i = 0; i < 4; ++i) {
        float4 t = xp[i];
        xv[4 * i + 0] = t.x; xv[4 * i + 1] = t.y;
        xv[4 * i + 2] = t.z; xv[4 * i + 3] = t.w;
    }

    // LN -> swish -> 16x16 matmul (edge_proj)
    float mu = 0.f;
#pragma unroll
    for (int i = 0; i < 16; ++i) mu += xv[i];
    mu *= (1.f / 16.f);
    float var = 0.f;
#pragma unroll
    for (int i = 0; i < 16; ++i) { float d = xv[i] - mu; var += d * d; }
    var *= (1.f / 16.f);
    float rs = rsqrtf(var + EPS);

    float fe[16];
#pragma unroll
    for (int j = 0; j < 16; ++j) fe[j] = sbias[j];
#pragma unroll
    for (int i = 0; i < 16; ++i) {
        float sw = swishf((xv[i] - mu) * rs * sg[i] + sb[i]);
#pragma unroll
        for (int j = 0; j < 16; ++j) fe[j] += sw * sW[i * 16 + j];
    }

    // masked scores, transposed to [B,H,N,M] bf16 for the attention kernel
    bool masked = (adj[(size_t)(b * N_DIM + n) * N_DIM + m] == 0);
    size_t sbase = ((size_t)(b * 16) * N_DIM + n) * N_DIM + m;
#pragma unroll
    for (int h = 0; h < 16; ++h) {
        float s = masked ? -1.0e9f : fe[h];
        score_ws[sbase + (size_t)h * N_DIM * N_DIM] = (bf16)s;
    }

    // edge_out = LN-swish(fe) @ eo_W + eo_bias + residual
    float mu2 = 0.f;
#pragma unroll
    for (int i = 0; i < 16; ++i) mu2 += fe[i];
    mu2 *= (1.f / 16.f);
    float var2 = 0.f;
#pragma unroll
    for (int i = 0; i < 16; ++i) { float d = fe[i] - mu2; var2 += d * d; }
    var2 *= (1.f / 16.f);
    float rs2 = rsqrtf(var2 + EPS);

    float o[16];
#pragma unroll
    for (int j = 0; j < 16; ++j) o[j] = sobias[j] + xv[j];
#pragma unroll
    for (int i = 0; i < 16; ++i) {
        float sw = swishf((fe[i] - mu2) * rs2 * sog[i] + sob[i]);
#pragma unroll
        for (int j = 0; j < 16; ++j) o[j] += sw * sWo[i * 16 + j];
    }
    float4* op = (float4*)(edges_out + e * 16);
#pragma unroll
    for (int i = 0; i < 4; ++i)
        op[i] = make_float4(o[4 * i], o[4 * i + 1], o[4 * i + 2], o[4 * i + 3]);
}

// -------------------------------------------------------------------------
// Kernel 2: LN->swish of node rows (128 wide) -> bf16. One wave per row.
// -------------------------------------------------------------------------
__global__ __launch_bounds__(128) void node_ln_kernel(
    const float* __restrict__ f_nodes,
    const float* __restrict__ g, const float* __restrict__ bta,
    bf16* __restrict__ xn)
{
    int tid = threadIdx.x;
    int lane = tid & 31;
    int row = blockIdx.x * 4 + (tid >> 5);
    const float4 x4 = ((const float4*)(f_nodes + (size_t)row * DN))[lane];
    float s = x4.x + x4.y + x4.z + x4.w;
    float s2 = x4.x * x4.x + x4.y * x4.y + x4.z * x4.z + x4.w * x4.w;
#pragma unroll
    for (int off = 1; off < 32; off <<= 1) {
        s  += __shfl_xor(s, off, 32);
        s2 += __shfl_xor(s2, off, 32);
    }
    float mu = s * (1.f / 128.f);
    float var = s2 * (1.f / 128.f) - mu * mu;
    float rs = rsqrtf(var + EPS);
    float xv[4] = {x4.x, x4.y, x4.z, x4.w};
#pragma unroll
    for (int j = 0; j < 4; ++j) {
        int c = lane * 4 + j;
        float y = swishf((xv[j] - mu) * rs * g[c] + bta[c]);
        xn[(size_t)row * DN + c] = (bf16)y;
    }
}

// Assemble a v16bf fragment from two contiguous 8-element bf16 runs
__device__ __forceinline__ v16bf make_frag(const bf16* lo, const bf16* hi)
{
    v8bf a = *(const v8bf*)lo;
    v8bf b = *(const v8bf*)hi;
    v16bf r;
#pragma unroll
    for (int e = 0; e < 8; ++e) { r[e] = a[e]; r[8 + e] = b[e]; }
    return r;
}

// -------------------------------------------------------------------------
// Kernel 3: node_proj GEMM  [4096,128](bf16) x Wt[2048,128](bf16).
// One wave computes 16 rows x 64 cols. Epilogue scatters the projected
// heads transposed to v_ws[B,H,D,N] bf16 so attention B-frags are linear.
// -------------------------------------------------------------------------
__global__ __launch_bounds__(32) void gemm_nodeproj_kernel(
    const bf16* __restrict__ xn, const bf16* __restrict__ Wt /* [2048,128] */,
    const float* __restrict__ bias, bf16* __restrict__ v_ws)
{
    int lane = threadIdx.x;
    int mr = lane & 15, h2 = lane >> 4;
    int row0 = blockIdx.y * 16;
    int c0 = blockIdx.x * 64;

    v8f acc[4] = {};
    const bf16* arow = xn + (size_t)(row0 + mr) * DN;
#pragma unroll
    for (int k0 = 0; k0 < DN; k0 += 32) {
        v16bf a = make_frag(arow + k0 + 8 * h2, arow + k0 + 16 + 8 * h2);
#pragma unroll
        for (int t = 0; t < 4; ++t) {
            const bf16* wrow = Wt + (size_t)(c0 + t * 16 + mr) * DN + k0 + 8 * h2;
            v16bf bm = make_frag(wrow, wrow + 16);
            acc[t] = __builtin_amdgcn_wmma_f32_16x16x32_bf16(
                false, a, false, bm, (short)0, acc[t], false, false);
        }
    }
    // Epilogue: C VGPR r -> (M = r + 8*h2, Ncol = mr)
#pragma unroll
    for (int t = 0; t < 4; ++t) {
#pragma unroll
        for (int r = 0; r < 8; ++r) {
            int m = r + 8 * h2;
            int col = c0 + t * 16 + mr;
            int grow = row0 + m;
            int bb = grow >> 9, nn = grow & 511;
            int hh = col >> 7, dd = col & 127;
            float val = acc[t][r] + bias[col];
            v_ws[((size_t)((bb * 16 + hh) * DN + dd)) * N_DIM + nn] = (bf16)val;
        }
    }
}

// -------------------------------------------------------------------------
// Kernel 4: softmax + attn@v. One wave per (b,h, 16-row tile).
// score_ws: [B,H,N,M] bf16 (pre-masked). v_ws: [B,H,D,N] bf16.
// Output agg_ws: [B,N,2048] f32 (col = h*128+d).
// -------------------------------------------------------------------------
__global__ __launch_bounds__(32) void attn_kernel(
    const bf16* __restrict__ score_ws, const bf16* __restrict__ v_ws,
    float* __restrict__ agg_ws)
{
    int lane = threadIdx.x;
    int r = lane & 15, h2 = lane >> 4;
    int bh = blockIdx.x;                 // b*16 + h
    int n0 = blockIdx.y * 16;
    int b = bh >> 4, h = bh & 15;

    const bf16* srow = score_ws + ((size_t)bh * N_DIM + n0 + r) * N_DIM;

    // Pass 1: row max / sum-of-exp, lanes l and l^16 split the row
    float rmax = -3.0e38f;
    for (int j = 0; j < 32; ++j) {
        v8bf s8 = *(const v8bf*)(srow + h2 * 256 + j * 8);
#pragma unroll
        for (int e = 0; e < 8; ++e) rmax = fmaxf(rmax, (float)s8[e]);
    }
    rmax = fmaxf(rmax, __shfl_xor(rmax, 16, 32));
    float rsum = 0.f;
    for (int j = 0; j < 32; ++j) {
        v8bf s8 = *(const v8bf*)(srow + h2 * 256 + j * 8);
#pragma unroll
        for (int e = 0; e < 8; ++e) rsum += __expf((float)s8[e] - rmax);
    }
    rsum += __shfl_xor(rsum, 16, 32);
    float rinv = 1.0f / rsum;

    // Pass 2: attn tile (16x32 bf16) x v tiles (32x16) via WMMA
    v8f acc[8] = {};
    const bf16* vb = v_ws + (size_t)bh * DN * N_DIM;
    for (int m0 = 0; m0 < N_DIM; m0 += 32) {
        __builtin_prefetch(srow + m0 + 64, 0, 0);
        v8bf alo = *(const v8bf*)(srow + m0 + 8 * h2);
        v8bf ahi = *(const v8bf*)(srow + m0 + 16 + 8 * h2);
        v16bf a;
#pragma unroll
        for (int e = 0; e < 8; ++e) {
            a[e]     = (bf16)(__expf((float)alo[e] - rmax) * rinv);
            a[8 + e] = (bf16)(__expf((float)ahi[e] - rmax) * rinv);
        }
#pragma unroll
        for (int t = 0; t < 8; ++t) {
            int d = t * 16 + r;
            const bf16* vcol = vb + (size_t)d * N_DIM + m0 + 8 * h2;
            v16bf bm = make_frag(vcol, vcol + 16);
            acc[t] = __builtin_amdgcn_wmma_f32_16x16x32_bf16(
                false, a, false, bm, (short)0, acc[t], false, false);
        }
    }
    // Epilogue: agg[b, n0+m, h*128 + col]
#pragma unroll
    for (int t = 0; t < 8; ++t) {
#pragma unroll
        for (int rr = 0; rr < 8; ++rr) {
            int m = rr + 8 * h2;
            int col = t * 16 + r;
            agg_ws[(size_t)(b * N_DIM + n0 + m) * PROJ + h * DN + col] = acc[t][rr];
        }
    }
}

// -------------------------------------------------------------------------
// Kernel 5: LN->swish of agg rows (2048 wide) -> bf16. One block per row.
// -------------------------------------------------------------------------
__global__ __launch_bounds__(256) void ln_agg_kernel(
    const float* __restrict__ agg, const float* __restrict__ g,
    const float* __restrict__ bta, bf16* __restrict__ aggn)
{
    __shared__ float smem[16];
    int tid = threadIdx.x;
    size_t row = blockIdx.x;
    const float4* x = (const float4*)(agg + row * PROJ);
    float4 xv[2];
    float s = 0.f, s2 = 0.f;
#pragma unroll
    for (int i = 0; i < 2; ++i) {
        xv[i] = x[tid * 2 + i];
        s  += xv[i].x + xv[i].y + xv[i].z + xv[i].w;
        s2 += xv[i].x * xv[i].x + xv[i].y * xv[i].y + xv[i].z * xv[i].z + xv[i].w * xv[i].w;
    }
#pragma unroll
    for (int off = 1; off < 32; off <<= 1) {
        s  += __shfl_xor(s, off, 32);
        s2 += __shfl_xor(s2, off, 32);
    }
    int wid = tid >> 5, lid = tid & 31;
    if (lid == 0) { smem[wid] = s; smem[8 + wid] = s2; }
    __syncthreads();
    if (tid == 0) {
        float ts = 0.f, ts2 = 0.f;
        for (int w = 0; w < 8; ++w) { ts += smem[w]; ts2 += smem[8 + w]; }
        smem[0] = ts; smem[8] = ts2;
    }
    __syncthreads();
    float mu = smem[0] * (1.f / 2048.f);
    float var = smem[8] * (1.f / 2048.f) - mu * mu;
    float rs = rsqrtf(var + EPS);
#pragma unroll
    for (int i = 0; i < 2; ++i) {
        float v4[4] = {xv[i].x, xv[i].y, xv[i].z, xv[i].w};
#pragma unroll
        for (int j = 0; j < 4; ++j) {
            int c = tid * 8 + i * 4 + j;
            float y = swishf((v4[j] - mu) * rs * g[c] + bta[c]);
            aggn[row * PROJ + c] = (bf16)y;
        }
    }
}

// -------------------------------------------------------------------------
// Kernel 6: node_out GEMM [4096,2048](bf16) x Wt[128,2048](bf16)
// + bias + f_nodes residual -> nodes_out (f32).
// -------------------------------------------------------------------------
__global__ __launch_bounds__(32) void gemm_nodeout_kernel(
    const bf16* __restrict__ aggn, const bf16* __restrict__ Wt /* [128,2048] */,
    const float* __restrict__ bias, const float* __restrict__ f_nodes,
    float* __restrict__ nodes_out)
{
    int lane = threadIdx.x;
    int mr = lane & 15, h2 = lane >> 4;
    int row0 = blockIdx.y * 16;
    int c0 = blockIdx.x * 64;

    v8f acc[4] = {};
    const bf16* arow = aggn + (size_t)(row0 + mr) * PROJ;
    for (int k0 = 0; k0 < PROJ; k0 += 32) {
        __builtin_prefetch(arow + k0 + 64, 0, 0);
        v16bf a = make_frag(arow + k0 + 8 * h2, arow + k0 + 16 + 8 * h2);
#pragma unroll
        for (int t = 0; t < 4; ++t) {
            const bf16* wrow = Wt + (size_t)(c0 + t * 16 + mr) * PROJ + k0 + 8 * h2;
            v16bf bm = make_frag(wrow, wrow + 16);
            acc[t] = __builtin_amdgcn_wmma_f32_16x16x32_bf16(
                false, a, false, bm, (short)0, acc[t], false, false);
        }
    }
#pragma unroll
    for (int t = 0; t < 4; ++t) {
#pragma unroll
        for (int r = 0; r < 8; ++r) {
            int m = r + 8 * h2;
            int col = c0 + t * 16 + mr;
            size_t grow = (size_t)(row0 + m);
            float val = acc[t][r] + bias[col] + f_nodes[grow * DN + col];
            nodes_out[grow * DN + col] = val;
        }
    }
}

// -------------------------------------------------------------------------
extern "C" void kernel_launch(void* const* d_in, const int* in_sizes, int n_in,
                              void* d_out, int out_size, void* d_ws, size_t ws_size,
                              hipStream_t stream)
{
    const float* f_nodes = (const float*)d_in[0];
    const float* f_edges = (const float*)d_in[1];
    const int*   adj     = (const int*)d_in[2];
    const float* np_g    = (const float*)d_in[3];
    const float* np_b    = (const float*)d_in[4];
    const float* np_W    = (const float*)d_in[5];
    const float* np_bias = (const float*)d_in[6];
    const float* ep_g    = (const float*)d_in[7];
    const float* ep_b    = (const float*)d_in[8];
    const float* ep_W    = (const float*)d_in[9];
    const float* ep_bias = (const float*)d_in[10];
    const float* no_g    = (const float*)d_in[11];
    const float* no_b    = (const float*)d_in[12];
    const float* no_W    = (const float*)d_in[13];
    const float* no_bias = (const float*)d_in[14];
    const float* eo_g    = (const float*)d_in[15];
    const float* eo_b    = (const float*)d_in[16];
    const float* eo_W    = (const float*)d_in[17];
    const float* eo_bias = (const float*)d_in[18];

    float* nodes_out = (float*)d_out;                          // 8*512*128
    float* edges_out = (float*)d_out + (size_t)ROWS * DN;      // 8*512*512*16

    // Workspace layout (~130 MB total)
    char* ws = (char*)d_ws;
    bf16*  score_ws = (bf16*)ws;                               //  64 MB [B,H,N,M]
    bf16*  v_ws     = (bf16*)(ws + ((size_t)64 << 20));        //  16 MB [B,H,D,N]
    bf16*  xn_ws    = (bf16*)(ws + ((size_t)80 << 20));        //   1 MB [4096,128]
    float* agg_ws   = (float*)(ws + ((size_t)81 << 20));       //  32 MB [4096,2048]
    bf16*  aggn_ws  = (bf16*)(ws + ((size_t)113 << 20));       //  16 MB [4096,2048]
    bf16*  wt_np    = (bf16*)(ws + ((size_t)129 << 20));       // 0.5 MB [2048,128]
    bf16*  wt_no    = (bf16*)(ws + ((size_t)129 << 20) + (512 << 10)); // 0.5 MB [128,2048]

    // 0. weight transpose+convert (np_W: K=128,C=2048 ; no_W: K=2048,C=128)
    transpose_w_kernel<<<dim3((DN * PROJ) / 256), dim3(256), 0, stream>>>(
        np_W, wt_np, 11, DN);
    transpose_w_kernel<<<dim3((PROJ * DN) / 256), dim3(256), 0, stream>>>(
        no_W, wt_no, 7, PROJ);

    // 1. fused edge pipeline (streams the big tensors once)
    edge_kernel<<<dim3((B_DIM * N_DIM * N_DIM) / 256), dim3(256), 0, stream>>>(
        f_edges, adj, ep_g, ep_b, ep_W, ep_bias, eo_g, eo_b, eo_W, eo_bias,
        score_ws, edges_out);

    // 2. node LN -> bf16
    node_ln_kernel<<<dim3(ROWS / 4), dim3(128), 0, stream>>>(
        f_nodes, np_g, np_b, xn_ws);

    // 3. node_proj GEMM -> v (transposed per-head)
    gemm_nodeproj_kernel<<<dim3(PROJ / 64, ROWS / 16), dim3(32), 0, stream>>>(
        xn_ws, wt_np, np_bias, v_ws);

    // 4. softmax + attn@v
    attn_kernel<<<dim3(B_DIM * 16, N_DIM / 16), dim3(32), 0, stream>>>(
        score_ws, v_ws, agg_ws);

    // 5. agg LN -> bf16
    ln_agg_kernel<<<dim3(ROWS), dim3(256), 0, stream>>>(
        agg_ws, no_g, no_b, aggn_ws);

    // 6. node_out GEMM + residual
    gemm_nodeout_kernel<<<dim3(DN / 64, ROWS / 16), dim3(32), 0, stream>>>(
        aggn_ws, wt_no, no_bias, f_nodes, nodes_out);
}
